// WeightedGCNLayer_88708254531814
// MI455X (gfx1250) — matrix operand order repaired and checked by reading, not compiled, not run on hardware
//
#include <hip/hip_runtime.h>
#include <hip/hip_bf16.h>

typedef __attribute__((ext_vector_type(16))) __bf16 v16bf;
typedef __attribute__((ext_vector_type(8)))  __bf16 v8bf;
typedef __attribute__((ext_vector_type(8)))  float  v8f;

#define NN    2048   // nodes
#define FIN   512
#define FOUT  512
#define BATCH 16

// ---------------- Stage 0: deg_inv_sqrt ----------------
__global__ void __launch_bounds__(256)
gcn_deg_kernel(const float* __restrict__ adj, float* __restrict__ dis) {
    __shared__ float red[256];
    const float* row = adj + (size_t)blockIdx.x * NN;
    float s = 0.f;
    for (int i = threadIdx.x; i < NN; i += 256) s += row[i];
    red[threadIdx.x] = s;
    __syncthreads();
    for (int off = 128; off > 0; off >>= 1) {
        if (threadIdx.x < off) red[threadIdx.x] += red[threadIdx.x + off];
        __syncthreads();
    }
    if (threadIdx.x == 0) {
        float d = red[0];
        dis[blockIdx.x] = (d > 0.f) ? rsqrtf(d) : 0.f;
    }
}

// ---------------- Stage 0b: Wt[o][k] = bf16(W[k][o]) ----------------
__global__ void __launch_bounds__(256)
gcn_wt_kernel(const float* __restrict__ W, __bf16* __restrict__ Wt) {
    int idx = blockIdx.x * 256 + threadIdx.x;   // o*FIN + k
    int o = idx / FIN, k = idx % FIN;
    Wt[idx] = (__bf16)W[(size_t)k * FOUT + o];
}

// ---------------- Stage 0c: norm[n][m] = bf16(adj*dis[n]*dis[m]) ----------------
__global__ void __launch_bounds__(256)
gcn_norm_kernel(const float* __restrict__ adj, const float* __restrict__ dis,
                __bf16* __restrict__ nrm) {
    int idx = blockIdx.x * 256 + threadIdx.x;
    int n = idx / NN, m = idx % NN;
    nrm[idx] = (__bf16)(adj[idx] * dis[n] * dis[m]);
}

// ===================================================================
// Stage 1: support_t[b][o][node] = bf16( x @ W )
// Workgroup tile: 128 (M = flattened b*NN+node) x 64 (N = FOUT).
// LDS-staged, K-step 32. 8 waves; wave w owns M-strip [w*16, w*16+16),
// sweeps 4 N-subtiles -> 4 WMMA per K-step per wave.
// ===================================================================
__global__ void __launch_bounds__(256)
gcn_gemm_xw_kernel(const float* __restrict__ x, const __bf16* __restrict__ Wt,
                   __bf16* __restrict__ st) {
    __shared__ __bf16 lA[128 * 32];   // [mrow][k]   8 KB
    __shared__ __bf16 lB[64 * 32];    // [ncol][k]   4 KB

    const int tid  = threadIdx.x;
    const int wave = tid >> 5;
    const int lane = tid & 31;
    const int llo  = lane & 15;
    const int hi   = lane >> 4;

    const int o0 = blockIdx.x * 64;     // FOUT tile
    const int m0 = blockIdx.y * 128;    // row tile (never crosses batch: 2048%128==0)

    // fill-phase indices
    const int frow  = tid >> 1;         // 0..127  (A: one 32B bf16 chunk = 16 f32)
    const int fhalf = tid & 1;          // 0..1
    const int brow  = tid >> 2;         // 0..63   (B: one 16B chunk)
    const int bq    = tid & 3;          // 0..3

    const float*  gA = x  + (size_t)(m0 + frow) * FIN;
    const __bf16* gB = Wt + (size_t)(o0 + brow) * FIN;

    v8f acc[4] = {};

    for (int kk = 0; kk < FIN; kk += 32) {
        // ---- cooperative fill: A (f32 -> bf16 in flight) ----
        {
            const float* src = gA + kk + fhalf * 16;
            float fa[16];
            *(float4*)(fa + 0)  = *(const float4*)(src + 0);
            *(float4*)(fa + 4)  = *(const float4*)(src + 4);
            *(float4*)(fa + 8)  = *(const float4*)(src + 8);
            *(float4*)(fa + 12) = *(const float4*)(src + 12);
            v16bf t;
#pragma unroll
            for (int i = 0; i < 16; ++i) t[i] = (__bf16)fa[i];
            *(v16bf*)(&lA[frow * 32 + fhalf * 16]) = t;
        }
        // ---- cooperative fill: B (already bf16) ----
        *(v8bf*)(&lB[brow * 32 + bq * 8]) = *(const v8bf*)(gB + kk + bq * 8);

        if (kk + 32 < FIN) __builtin_prefetch(gA + kk + 32, 0, 1);
        __syncthreads();

        // ---- compute: 1 A-frag, 4 B-frags, 4 WMMA ----
        const int ka = hi ? 16 : 0;
        const int kb = hi ? 16 : 0;
        const int arow = wave * 16 + llo;
        v8bf a0 = *(const v8bf*)(&lA[arow * 32 + ka]);
        v8bf a1 = *(const v8bf*)(&lA[arow * 32 + 8 + ka]);
        v16bf a;
#pragma unroll
        for (int i = 0; i < 8; ++i) { a[i] = a0[i]; a[i + 8] = a1[i]; }
#pragma unroll
        for (int ns = 0; ns < 4; ++ns) {
            v16bf bm = *(const v16bf*)(&lB[(ns * 16 + llo) * 32 + kb]);
            acc[ns] = __builtin_amdgcn_wmma_f32_16x16x32_bf16(
                false, a, false, bm, (short)0, acc[ns], false, false);
        }
        __syncthreads();
    }

    // ---- epilogue: transposed bf16 store (D: VGPR r -> M=r+8*hi, lane llo -> N)
    const int bidx  = m0 >> 11;
    const int node0 = m0 - (bidx << 11) + wave * 16 + (hi ? 8 : 0);
#pragma unroll
    for (int ns = 0; ns < 4; ++ns) {
        __bf16* dst = st + ((size_t)bidx * FOUT + (o0 + ns * 16 + llo)) * NN + node0;
        v8bf o;
#pragma unroll
        for (int r = 0; r < 8; ++r) o[r] = (__bf16)acc[ns][r];
        *(v8bf*)dst = o;
    }
}

// ===================================================================
// Stage 2: out[b][n][o] = norm @ support + bias
// Workgroup tile: 128 (n) x 64 (o), K = 2048 over nodes, LDS-staged.
// ===================================================================
__global__ void __launch_bounds__(256)
gcn_gemm_adj_kernel(const __bf16* __restrict__ nrm, const __bf16* __restrict__ st,
                    const float* __restrict__ bias, float* __restrict__ out) {
    __shared__ __bf16 lA[128 * 32];   // [nrow][k]  8 KB
    __shared__ __bf16 lB[64 * 32];    // [ocol][k]  4 KB

    const int tid  = threadIdx.x;
    const int wave = tid >> 5;
    const int lane = tid & 31;
    const int llo  = lane & 15;
    const int hi   = lane >> 4;

    const int o0 = blockIdx.x * 64;     // FOUT tile
    const int n0 = blockIdx.y * 128;    // node tile
    const int b  = blockIdx.z;          // batch

    const int frow  = tid >> 1;         // 0..127
    const int fhalf = tid & 1;
    const int brow  = tid >> 2;         // 0..63
    const int bq    = tid & 3;

    const __bf16* gA = nrm + (size_t)(n0 + frow) * NN;
    const __bf16* gB = st  + ((size_t)b * FOUT + (o0 + brow)) * NN;

    v8f acc[4] = {};

    for (int kk = 0; kk < NN; kk += 32) {
        // ---- cooperative fills (both operands already bf16) ----
        *(v16bf*)(&lA[frow * 32 + fhalf * 16]) =
            *(const v16bf*)(gA + kk + fhalf * 16);
        *(v8bf*)(&lB[brow * 32 + bq * 8]) =
            *(const v8bf*)(gB + kk + bq * 8);

        if (kk + 32 < NN) __builtin_prefetch(gA + kk + 32, 0, 1);
        __syncthreads();

        // ---- compute ----
        const int ka = hi ? 16 : 0;
        const int kb = hi ? 16 : 0;
        const int arow = wave * 16 + llo;
        v8bf a0 = *(const v8bf*)(&lA[arow * 32 + ka]);
        v8bf a1 = *(const v8bf*)(&lA[arow * 32 + 8 + ka]);
        v16bf a;
#pragma unroll
        for (int i = 0; i < 8; ++i) { a[i] = a0[i]; a[i + 8] = a1[i]; }
#pragma unroll
        for (int ns = 0; ns < 4; ++ns) {
            v16bf bm = *(const v16bf*)(&lB[(ns * 16 + llo) * 32 + kb]);
            acc[ns] = __builtin_amdgcn_wmma_f32_16x16x32_bf16(
                false, a, false, bm, (short)0, acc[ns], false, false);
        }
        __syncthreads();
    }

    // ---- epilogue: f32 store + bias ----
    const int nrow = n0 + wave * 16 + (hi ? 8 : 0);
#pragma unroll
    for (int ns = 0; ns < 4; ++ns) {
        const int oc = o0 + ns * 16 + llo;
        const float bv = bias[oc];
        float* dst = out + ((size_t)b * NN + nrow) * FOUT + oc;
#pragma unroll
        for (int r = 0; r < 8; ++r) dst[(size_t)r * FOUT] = acc[ns][r] + bv;
    }
}

// ---------------- launch ----------------
extern "C" void kernel_launch(void* const* d_in, const int* in_sizes, int n_in,
                              void* d_out, int out_size, void* d_ws, size_t ws_size,
                              hipStream_t stream) {
    const float* x    = (const float*)d_in[0];   // [16,2048,512]
    const float* adj  = (const float*)d_in[1];   // [2048,2048]
    const float* W    = (const float*)d_in[2];   // [512,512]
    const float* bias = (const float*)d_in[3];   // [512]
    float* out = (float*)d_out;                  // [16,2048,512]

    char* ws = (char*)d_ws;
    float*  dis = (float*)ws;                                     //   8 KB
    __bf16* Wt  = (__bf16*)(ws + 8 * 1024);                       // 512 KB
    __bf16* nrm = (__bf16*)(ws + 8 * 1024 + 512 * 1024);          //   8 MB
    __bf16* st  = (__bf16*)(ws + 8 * 1024 + 512 * 1024
                               + (size_t)8 * 1024 * 1024);        //  32 MB

    gcn_deg_kernel<<<NN, 256, 0, stream>>>(adj, dis);
    gcn_wt_kernel<<<(FIN * FOUT) / 256, 256, 0, stream>>>(W, Wt);
    gcn_norm_kernel<<<(NN * NN) / 256, 256, 0, stream>>>(adj, dis, nrm);

    dim3 g1(FOUT / 64, (BATCH * NN) / 128);          // 8 x 256
    gcn_gemm_xw_kernel<<<g1, 256, 0, stream>>>(x, Wt, st);

    dim3 g2(FOUT / 64, NN / 128, BATCH);             // 8 x 16 x 16
    gcn_gemm_adj_kernel<<<g2, 256, 0, stream>>>(nrm, st, bias, out);
}
